// WHISPER_DECODER_10273561772671
// MI455X (gfx1250) — matrix operand order repaired and checked by reading, not compiled, not run on hardware
//
#include <hip/hip_runtime.h>
#include <hip/hip_bf16.h>

// Whisper decoder forward on gfx1250.
// All GEMMs: bf16 operands (pre-converted / pre-transposed), f32 accumulate via
// v_wmma_f32_16x16x32_bf16, 32x32 tile per wave, guard-free inner loop.

#define NL   12
#define NH   12
#define DM   768
#define HDM  64
#define FFN_ 3072
#define VOC  51865
#define SEQ  448
#define ENCL 1500
#define LDSC 1504   // padded K/width for cross-attention (multiple of 32)

typedef __attribute__((ext_vector_type(16))) __bf16         v16bf;
typedef __attribute__((ext_vector_type(8)))  float          v8f;
typedef __attribute__((ext_vector_type(8)))  unsigned short us8;

enum { FLAG_GELU = 1, FLAG_TRANS = 2, FLAG_BTRANS = 4 };

__device__ __forceinline__ unsigned short f2bf(float f) {
  unsigned int u = __builtin_bit_cast(unsigned int, f);
  u += 0x7fffu + ((u >> 16) & 1u);          // round-to-nearest-even
  return (unsigned short)(u >> 16);
}

union BfFrag { v16bf v; us8 h[2]; };

// ---------------------------------------------------------------------------
// GEMM: C = act(A * Bt^T + bias) (+residual).  A: bf16 [M x K] row-major.
// Bt: bf16 [N x K] (pre-transposed B).  One wave computes a 32x32 tile.
// K must be a multiple of 32 (callers pad).  N may be ragged (clamped loads,
// guarded stores).  Optional f32 output (plain or transposed) and bf16 output
// (plain with own ld / per-batch column offset, or transposed contiguous).
// ---------------------------------------------------------------------------
__global__ __launch_bounds__(32)
void gemm_bf16_wmma(const unsigned short* __restrict__ A,
                    const unsigned short* __restrict__ Bt,
                    const float* __restrict__ bias, const float* __restrict__ residual,
                    float* __restrict__ Cf, unsigned short* __restrict__ Cbf,
                    int M, int N, int K, int lda, int ldbt, int ldc, int ldbf,
                    long long aB, long long bB, long long cB, long long cbfB,
                    long long biasB, int flags) {
  const int lane = threadIdx.x & 31;
  const int half = lane >> 4;
  const int l16  = lane & 15;
  const int m0 = blockIdx.y * 32;
  const int n0 = blockIdx.x * 32;
  const long long bz = blockIdx.z;

  A  += bz * aB;
  Bt += bz * bB;
  const float*    bias_p = bias     ? bias + bz * biasB : nullptr;
  const float*    res_p  = residual ? residual + bz * cB : nullptr;
  float*          cf     = Cf       ? Cf + bz * cB : nullptr;
  unsigned short* cb     = Cbf      ? Cbf + bz * cbfB : nullptr;

  const unsigned short* a0 = A + (long long)(m0 + l16) * lda + half * 8;
  const unsigned short* a1 = a0 + (long long)16 * lda;
  const int bn0 = n0 + l16, bn1 = bn0 + 16;
  const int bn0c = bn0 < N ? bn0 : N - 1;
  const int bn1c = bn1 < N ? bn1 : N - 1;
  const unsigned short* b0 = Bt + (long long)bn0c * ldbt + half * 8;
  const unsigned short* b1 = Bt + (long long)bn1c * ldbt + half * 8;

  v8f acc00 = {}, acc01 = {}, acc10 = {}, acc11 = {};
  for (int k0 = 0; k0 < K; k0 += 32) {
    if (k0 + 32 < K) {                    // next-tile prefetch -> global_prefetch_b8
      __builtin_prefetch(a0 + k0 + 32, 0, 0);
      __builtin_prefetch(b0 + k0 + 32, 0, 0);
    }
    BfFrag fa0, fa1, fb0, fb1;
    fa0.h[0] = *(const us8*)(a0 + k0);        // A 16x32 frag: K half*8+0..7 | +16..23
    fa0.h[1] = *(const us8*)(a0 + k0 + 16);
    fa1.h[0] = *(const us8*)(a1 + k0);
    fa1.h[1] = *(const us8*)(a1 + k0 + 16);
    fb0.h[0] = *(const us8*)(b0 + k0);        // B 32x16 frag: K half*8+0..7 | +16..23
    fb0.h[1] = *(const us8*)(b0 + k0 + 16);
    fb1.h[0] = *(const us8*)(b1 + k0);
    fb1.h[1] = *(const us8*)(b1 + k0 + 16);
    acc00 = __builtin_amdgcn_wmma_f32_16x16x32_bf16(false, fa0.v, false, fb0.v, (short)0, acc00, false, false);
    acc01 = __builtin_amdgcn_wmma_f32_16x16x32_bf16(false, fa0.v, false, fb1.v, (short)0, acc01, false, false);
    acc10 = __builtin_amdgcn_wmma_f32_16x16x32_bf16(false, fa1.v, false, fb0.v, (short)0, acc10, false, false);
    acc11 = __builtin_amdgcn_wmma_f32_16x16x32_bf16(false, fa1.v, false, fb1.v, (short)0, acc11, false, false);
  }

  auto emit = [&](const v8f& acc, int mbase, int bn, int bnc, bool nOK) {
    const float bval = bias_p ? bias_p[bnc] : 0.0f;
    unsigned short pk[8];
#pragma unroll
    for (int r = 0; r < 8; ++r) {
      const int m = mbase + r;
      float v = acc[r] + bval;
      if (flags & FLAG_GELU) v = 0.5f * v * (1.0f + erff(v * 0.70710678118654752f));
      if (cf && nOK) {
        if (flags & FLAG_TRANS) {
          cf[(long long)bn * ldc + m] = v;                 // K-cache [HD, S]
        } else {
          const long long idx = (long long)m * ldc + bn;
          float o = v;
          if (res_p) o += res_p[idx];
          cf[idx] = o;
        }
      }
      if (cb) {
        if (flags & FLAG_BTRANS) pk[r] = f2bf(v);
        else if (nOK) cb[(long long)m * ldbf + bn] = f2bf(v);
      }
    }
    if (cb && (flags & FLAG_BTRANS) && nOK) {
      us8 p;
#pragma unroll
      for (int r = 0; r < 8; ++r) p[r] = pk[r];
      *(us8*)(cb + (long long)bn * ldbf + mbase) = p;      // contiguous 16B store
    }
  };
  emit(acc00, m0 + half * 8,      bn0, bn0c, bn0 < N);
  emit(acc10, m0 + 16 + half * 8, bn0, bn0c, bn0 < N);
  emit(acc01, m0 + half * 8,      bn1, bn1c, bn1 < N);
  emit(acc11, m0 + 16 + half * 8, bn1, bn1c, bn1 < N);
}

// ---------------------------------------------------------------------------
// Transpose-convert: f32 [K x N] (ldin) -> bf16 [N x Kpad] (ldout), zero-pad
// K..Kpad.  LDS 32x32 tile for coalescing; grid.z batches (per-head weights).
// ---------------------------------------------------------------------------
__global__ __launch_bounds__(256)
void cvt_tr_kernel(const float* __restrict__ in, unsigned short* __restrict__ out,
                   int K, int N, int ldin, int ldout, long long inB, long long outB,
                   int Kpad) {
  __shared__ float tile[32][33];
  in  += (long long)blockIdx.z * inB;
  out += (long long)blockIdx.z * outB;
  const int k0 = blockIdx.y * 32, n0 = blockIdx.x * 32;
  const int tx = threadIdx.x & 31, ty = threadIdx.x >> 5;
  for (int yy = ty; yy < 32; yy += 8) {
    const int k = k0 + yy, n = n0 + tx;
    tile[yy][tx] = (k < K && n < N) ? in[(long long)k * ldin + n] : 0.0f;
  }
  __syncthreads();
  for (int yy = ty; yy < 32; yy += 8) {
    const int n = n0 + yy, k = k0 + tx;
    if (n < N && k < Kpad) out[(long long)n * ldout + k] = f2bf(tile[tx][yy]);
  }
}

// ---------------------------------------------------------------------------
// Row LayerNorm over DM; writes f32 and/or bf16 (nullable).
// ---------------------------------------------------------------------------
__global__ __launch_bounds__(256)
void ln_kernel(const float* __restrict__ x, const float* __restrict__ g,
               const float* __restrict__ b, float* __restrict__ yf,
               unsigned short* __restrict__ yb) {
  __shared__ float red[256];
  const int row = blockIdx.x;
  const int t = threadIdx.x;
  const float* xr = x + (long long)row * DM;
  float s = 0.0f;
  for (int i = t; i < DM; i += 256) s += xr[i];
  red[t] = s; __syncthreads();
  for (int o = 128; o > 0; o >>= 1) { if (t < o) red[t] += red[t + o]; __syncthreads(); }
  const float mu = red[0] * (1.0f / DM);
  __syncthreads();
  float v = 0.0f;
  for (int i = t; i < DM; i += 256) { float d0 = xr[i] - mu; v += d0 * d0; }
  red[t] = v; __syncthreads();
  for (int o = 128; o > 0; o >>= 1) { if (t < o) red[t] += red[t + o]; __syncthreads(); }
  const float rs = rsqrtf(red[0] * (1.0f / DM) + 1e-5f);
  for (int i = t; i < DM; i += 256) {
    const float o = (xr[i] - mu) * rs * g[i] + b[i];
    if (yf) yf[(long long)row * DM + i] = o;
    if (yb) yb[(long long)row * DM + i] = f2bf(o);
  }
}

// ---------------------------------------------------------------------------
// Softmax over W columns of f32 scores; writes bf16 probabilities, zero-padded
// to Wpad.  Optional causal mask (-128 * mask_flag for col > row).
// grid = (rows, heads); f32 buffer reused as exp scratch.
// ---------------------------------------------------------------------------
__global__ __launch_bounds__(256)
void softmax_kernel(float* __restrict__ sc, unsigned short* __restrict__ pb,
                    const float* __restrict__ mflag, int W, int Wpad, int ld,
                    long long bstride, int causal) {
  __shared__ float red[256];
  const int row = blockIdx.x;
  float*          xr = sc + (long long)blockIdx.y * bstride + (long long)row * ld;
  unsigned short* pr = pb + (long long)blockIdx.y * bstride + (long long)row * ld;
  const float mf = mflag[0];
  const int t = threadIdx.x;
  float mx = -3.4e38f;
  for (int j = t; j < W; j += 256) {
    const float v = xr[j] + ((causal && j > row) ? -128.0f * mf : 0.0f);
    mx = fmaxf(mx, v);
  }
  red[t] = mx; __syncthreads();
  for (int o = 128; o > 0; o >>= 1) { if (t < o) red[t] = fmaxf(red[t], red[t + o]); __syncthreads(); }
  mx = red[0]; __syncthreads();
  float s = 0.0f;
  for (int j = t; j < W; j += 256) {
    const float v = xr[j] + ((causal && j > row) ? -128.0f * mf : 0.0f);
    const float e = expf(v - mx);
    xr[j] = e;
    s += e;
  }
  red[t] = s; __syncthreads();
  for (int o = 128; o > 0; o >>= 1) { if (t < o) red[t] += red[t + o]; __syncthreads(); }
  const float inv = 1.0f / red[0];
  for (int j = t; j < Wpad; j += 256)
    pr[j] = (j < W) ? f2bf(xr[j] * inv) : (unsigned short)0;
}

__global__ __launch_bounds__(256)
void embed_kernel(const int* __restrict__ ids, const float* __restrict__ tok,
                  const float* __restrict__ pos, const int* __restrict__ hist,
                  float* __restrict__ h) {
  const int s = blockIdx.x;
  const long long id = ids[s];
  const long long p = hist[0] + s;
  for (int d0 = threadIdx.x; d0 < DM; d0 += 256)
    h[(long long)s * DM + d0] = tok[id * DM + d0] + pos[p * DM + d0];
}

// logits[v] = dot(hl, tok[v]) + suppress[v]; one wave32 per vocab row.
__global__ __launch_bounds__(256)
void logits_kernel(const float* __restrict__ hl, const float* __restrict__ tok,
                   const float* __restrict__ sup, float* __restrict__ outL) {
  const int lane = threadIdx.x & 31;
  const int rowv = blockIdx.x * 8 + (threadIdx.x >> 5);
  if (rowv >= VOC) return;
  const float* w = tok + (long long)rowv * DM;
  float s = 0.0f;
  for (int j = lane; j < DM; j += 32) s += hl[j] * w[j];
  for (int m = 16; m; m >>= 1) s += __shfl_xor(s, m, 32);
  if (lane == 0) outL[rowv] = s + sup[rowv];
}

__global__ __launch_bounds__(256)
void argmax_kernel(const float* __restrict__ x, int* __restrict__ outIdx) {
  __shared__ float bv[256];
  __shared__ int   bi[256];
  const int t = threadIdx.x;
  float best = -3.4e38f; int bidx = 0;
  for (int v = t; v < VOC; v += 256) {
    const float f = x[v];
    if (f > best) { best = f; bidx = v; }
  }
  bv[t] = best; bi[t] = bidx; __syncthreads();
  for (int o = 128; o > 0; o >>= 1) {
    if (t < o && (bv[t + o] > bv[t] || (bv[t + o] == bv[t] && bi[t + o] < bi[t]))) {
      bv[t] = bv[t + o]; bi[t] = bi[t + o];
    }
    __syncthreads();
  }
  if (t == 0) outIdx[0] = bi[0];
}

// ---------------------------------------------------------------------------
extern "C" void kernel_launch(void* const* d_in, const int* in_sizes, int n_in,
                              void* d_out, int out_size, void* d_ws, size_t ws_size,
                              hipStream_t stream) {
  (void)in_sizes; (void)n_in; (void)out_size; (void)ws_size;
  const int*   ids       = (const int*)d_in[0];
  const float* cross_k   = (const float*)d_in[3];
  const float* cross_v   = (const float*)d_in[4];
  const float* mask_flag = (const float*)d_in[5];
  const float* tok       = (const float*)d_in[6];
  const float* pos       = (const float*)d_in[7];
  const float* sa_ln_g   = (const float*)d_in[8];
  const float* sa_ln_b   = (const float*)d_in[9];
  const float* sa_qw     = (const float*)d_in[10];
  const float* sa_qb     = (const float*)d_in[11];
  const float* sa_kw     = (const float*)d_in[12];
  const float* sa_vw     = (const float*)d_in[13];
  const float* sa_vb     = (const float*)d_in[14];
  const float* sa_ow     = (const float*)d_in[15];
  const float* sa_ob     = (const float*)d_in[16];
  const float* ca_ln_g   = (const float*)d_in[17];
  const float* ca_ln_b   = (const float*)d_in[18];
  const float* ca_qw     = (const float*)d_in[19];
  const float* ca_qb     = (const float*)d_in[20];
  const float* ca_ow     = (const float*)d_in[21];
  const float* ca_ob     = (const float*)d_in[22];
  const float* fln_g     = (const float*)d_in[23];
  const float* fln_b     = (const float*)d_in[24];
  const float* fc1_w     = (const float*)d_in[25];
  const float* fc1_b     = (const float*)d_in[26];
  const float* fc2_w     = (const float*)d_in[27];
  const float* fc2_b     = (const float*)d_in[28];
  const float* f_ln_g    = (const float*)d_in[29];
  const float* f_ln_b    = (const float*)d_in[30];
  const float* sup       = (const float*)d_in[31];
  const int*   hist      = (const int*)d_in[32];

  // Outputs: new_k [L,H,HD,S] | new_v [L,H,S,HD] | logits [VOC] | max_id (i32)
  float* out    = (float*)d_out;
  float* outK   = out;
  float* outV   = outK + (size_t)NL * NH * HDM * SEQ;
  float* outLog = outV + (size_t)NL * NH * SEQ * HDM;
  int*   outMax = (int*)(outLog + VOC);

  // Workspace: f32 region, then bf16 (ushort) region.
  float* h  = (float*)d_ws;                                  // [SEQ, DM]
  float* sc = h + (size_t)SEQ * DM;                          // [NH, SEQ, LDSC]
  float* hl = sc + (size_t)NH * SEQ * LDSC;                  // [DM]
  unsigned short* hn_bf     = (unsigned short*)(hl + DM);    // [SEQ, DM]
  unsigned short* q_bf      = hn_bf     + (size_t)SEQ * DM;       // [NH, SEQ, HDM]
  unsigned short* k_bf      = q_bf      + (size_t)NH * SEQ * HDM; // [NH, SEQ, HDM]  (= Bt for scores)
  unsigned short* vT_bf     = k_bf      + (size_t)NH * SEQ * HDM; // [NH, HDM, SEQ]  (= Bt for attn@V)
  unsigned short* probs_bf  = vT_bf     + (size_t)NH * HDM * SEQ; // [NH, SEQ, LDSC]
  unsigned short* packed_bf = probs_bf  + (size_t)NH * SEQ * LDSC;// [SEQ, DM]
  unsigned short* ff_bf     = packed_bf + (size_t)SEQ * DM;       // [SEQ, FFN]
  unsigned short* wBt       = ff_bf     + (size_t)SEQ * FFN_;     // max FFN*DM

  auto gemm = [&](const unsigned short* A, const unsigned short* Bt, const float* bias,
                  const float* res, float* Cf, unsigned short* Cbf,
                  int M, int N, int K, int lda, int ldbt, int ldc, int ldbf,
                  long long aB, long long bB, long long cB, long long cbfB,
                  long long biasB, int batch, int flags) {
    dim3 grid((N + 31) / 32, M / 32, batch);
    gemm_bf16_wmma<<<grid, dim3(32), 0, stream>>>(A, Bt, bias, res, Cf, Cbf, M, N, K,
                                                  lda, ldbt, ldc, ldbf, aB, bB, cB, cbfB,
                                                  biasB, flags);
  };
  auto cvt_tr = [&](const float* in, unsigned short* o, int K, int N, int ldin, int ldout,
                    long long inB, long long outB, int Kpad, int batch) {
    dim3 grid((N + 31) / 32, (Kpad + 31) / 32, batch);
    cvt_tr_kernel<<<grid, dim3(256), 0, stream>>>(in, o, K, N, ldin, ldout, inB, outB, Kpad);
  };

  embed_kernel<<<SEQ, 256, 0, stream>>>(ids, tok, pos, hist, h);

  for (int i = 0; i < NL; ++i) {
    float* kOut = outK + (size_t)i * NH * HDM * SEQ;
    float* vOut = outV + (size_t)i * NH * SEQ * HDM;

    // ---------------- self attention ----------------
    ln_kernel<<<SEQ, 256, 0, stream>>>(h, sa_ln_g + i * DM, sa_ln_b + i * DM, nullptr, hn_bf);
    // Q: per-head [D x HD] weights -> Bt [HD x D]
    cvt_tr(sa_qw + (size_t)i * NH * DM * HDM, wBt, DM, HDM, HDM, DM,
           (long long)DM * HDM, (long long)HDM * DM, DM, NH);
    gemm(hn_bf, wBt, sa_qb + (size_t)i * NH * HDM, nullptr, nullptr, q_bf,
         SEQ, HDM, DM, DM, DM, 0, HDM,
         0, (long long)HDM * DM, 0, (long long)SEQ * HDM, HDM, NH, 0);
    // K: f32 transposed to KV-cache output + bf16 [S,HD] copy (Bt for scores)
    cvt_tr(sa_kw + (size_t)i * NH * DM * HDM, wBt, DM, HDM, HDM, DM,
           (long long)DM * HDM, (long long)HDM * DM, DM, NH);
    gemm(hn_bf, wBt, nullptr, nullptr, kOut, k_bf,
         SEQ, HDM, DM, DM, DM, SEQ, HDM,
         0, (long long)HDM * DM, (long long)HDM * SEQ, (long long)SEQ * HDM, 0, NH, FLAG_TRANS);
    // V: f32 [S,HD] to KV-cache output + bf16 transposed [HD,S] copy (Bt for attn@V)
    cvt_tr(sa_vw + (size_t)i * NH * DM * HDM, wBt, DM, HDM, HDM, DM,
           (long long)DM * HDM, (long long)HDM * DM, DM, NH);
    gemm(hn_bf, wBt, sa_vb + (size_t)i * NH * HDM, nullptr, vOut, vT_bf,
         SEQ, HDM, DM, DM, DM, HDM, SEQ,
         0, (long long)HDM * DM, (long long)SEQ * HDM, (long long)HDM * SEQ, HDM, NH, FLAG_BTRANS);
    // scores = q k^T  -> f32 [S x S] (ld LDSC)
    gemm(q_bf, k_bf, nullptr, nullptr, sc, nullptr,
         SEQ, SEQ, HDM, HDM, HDM, LDSC, 0,
         (long long)SEQ * HDM, (long long)SEQ * HDM, (long long)SEQ * LDSC, 0, 0, NH, 0);
    softmax_kernel<<<dim3(SEQ, NH), 256, 0, stream>>>(sc, probs_bf, mask_flag, SEQ, SEQ,
                                                      LDSC, (long long)SEQ * LDSC, 1);
    // attn = probs @ v  -> bf16 straight into packed [S, H*HD] (per-head column offset)
    gemm(probs_bf, vT_bf, nullptr, nullptr, nullptr, packed_bf,
         SEQ, HDM, SEQ, LDSC, SEQ, 0, DM,
         (long long)SEQ * LDSC, (long long)HDM * SEQ, 0, (long long)HDM, 0, NH, 0);
    // O-projection + residual
    cvt_tr(sa_ow + (size_t)i * DM * DM, wBt, DM, DM, DM, DM, 0, 0, DM, 1);
    gemm(packed_bf, wBt, sa_ob + (size_t)i * DM, h, h, nullptr,
         SEQ, DM, DM, DM, DM, DM, 0, 0, 0, 0, 0, 0, 1, 0);

    // ---------------- cross attention ----------------
    ln_kernel<<<SEQ, 256, 0, stream>>>(h, ca_ln_g + i * DM, ca_ln_b + i * DM, nullptr, hn_bf);
    cvt_tr(ca_qw + (size_t)i * NH * DM * HDM, wBt, DM, HDM, HDM, DM,
           (long long)DM * HDM, (long long)HDM * DM, DM, NH);
    gemm(hn_bf, wBt, ca_qb + (size_t)i * NH * HDM, nullptr, nullptr, q_bf,
         SEQ, HDM, DM, DM, DM, 0, HDM,
         0, (long long)HDM * DM, 0, (long long)SEQ * HDM, HDM, NH, 0);
    // cross_k [HD x ENC] -> Bt [ENC x HD]
    cvt_tr(cross_k + (size_t)i * NH * HDM * ENCL, wBt, HDM, ENCL, ENCL, HDM,
           (long long)HDM * ENCL, (long long)ENCL * HDM, HDM, NH);
    gemm(q_bf, wBt, nullptr, nullptr, sc, nullptr,
         SEQ, ENCL, HDM, HDM, HDM, LDSC, 0,
         (long long)SEQ * HDM, (long long)ENCL * HDM, (long long)SEQ * LDSC, 0, 0, NH, 0);
    softmax_kernel<<<dim3(SEQ, NH), 256, 0, stream>>>(sc, probs_bf, mask_flag, ENCL, LDSC,
                                                      LDSC, (long long)SEQ * LDSC, 0);
    // cross_v [ENC x HD] -> Bt [HD x 1504] zero-padded; K loop runs to 1504
    cvt_tr(cross_v + (size_t)i * NH * ENCL * HDM, wBt, ENCL, HDM, HDM, LDSC,
           (long long)ENCL * HDM, (long long)HDM * LDSC, LDSC, NH);
    gemm(probs_bf, wBt, nullptr, nullptr, nullptr, packed_bf,
         SEQ, HDM, LDSC, LDSC, LDSC, 0, DM,
         (long long)SEQ * LDSC, (long long)HDM * LDSC, 0, (long long)HDM, 0, NH, 0);
    cvt_tr(ca_ow + (size_t)i * DM * DM, wBt, DM, DM, DM, DM, 0, 0, DM, 1);
    gemm(packed_bf, wBt, ca_ob + (size_t)i * DM, h, h, nullptr,
         SEQ, DM, DM, DM, DM, DM, 0, 0, 0, 0, 0, 0, 1, 0);

    // ---------------- FFN ----------------
    ln_kernel<<<SEQ, 256, 0, stream>>>(h, fln_g + i * DM, fln_b + i * DM, nullptr, hn_bf);
    cvt_tr(fc1_w + (size_t)i * DM * FFN_, wBt, DM, FFN_, FFN_, DM, 0, 0, DM, 1);
    gemm(hn_bf, wBt, fc1_b + (size_t)i * FFN_, nullptr, nullptr, ff_bf,
         SEQ, FFN_, DM, DM, DM, 0, FFN_, 0, 0, 0, 0, 0, 1, FLAG_GELU);
    cvt_tr(fc2_w + (size_t)i * FFN_ * DM, wBt, FFN_, DM, DM, FFN_, 0, 0, FFN_, 1);
    gemm(ff_bf, wBt, fc2_b + (size_t)i * DM, h, h, nullptr,
         SEQ, DM, FFN_, FFN_, FFN_, DM, 0, 0, 0, 0, 0, 0, 1, 0);
  }

  // final LN on last row -> tied logits -> argmax
  ln_kernel<<<1, 256, 0, stream>>>(h + (size_t)(SEQ - 1) * DM, f_ln_g, f_ln_b, hl, nullptr);
  logits_kernel<<<(VOC + 7) / 8, 256, 0, stream>>>(hl, tok, sup, outLog);
  argmax_kernel<<<1, 256, 0, stream>>>(outLog, outMax);
}
